// NuSelectorModel_53506702573821
// MI455X (gfx1250) — matrix operand order repaired and checked by reading, not compiled, not run on hardware
//
#include <hip/hip_runtime.h>
#include <stdint.h>

#define BATCHN  32
#define SEQLEN  2048
#define DMODEL  128
#define DINNER  256
#define DSTATE  16
#define DTRANK  8
#define OUTPUTS 128
#define ML      (BATCHN * SEQLEN)   // 65536 GEMM rows
#define DBC_LD  48                  // 40 padded to 3 WMMA n-tiles
#define CHUNK   64                  // scan LDS staging depth

typedef __attribute__((ext_vector_type(16))) __bf16 bf16x16;
typedef __attribute__((ext_vector_type(8)))  float  f32x8;

struct alignas(16) U128 { uint32_t x, y, z, w; };
union FragU { U128 u[2]; bf16x16 v; };

__device__ __forceinline__ unsigned short f2bf(float f) {
  union { float f; uint32_t u; } x{f};
  uint32_t r = x.u + 0x7FFFu + ((x.u >> 16) & 1u);   // RNE
  return (unsigned short)(r >> 16);
}
__device__ __forceinline__ float bf2f(unsigned short h) {
  union { uint32_t u; float f; } x{(uint32_t)h << 16};
  return x.f;
}
__device__ __forceinline__ float sigm(float x) { return 1.0f / (1.0f + __expf(-x)); }

// ---- CDNA5 async global->LDS copy (ASYNCcnt path, no VGPR data movement) ----
// VDST = wave-relative LDS byte address (= low 32 bits of generic pointer per
// the LDS aperture mapping), VADDR = 64-bit global address.
__device__ __forceinline__ void async_g2l_b128(const void* gptr, const void* lptr) {
  uint32_t loff = (uint32_t)(uintptr_t)lptr;
  uint64_t g    = (uint64_t)(uintptr_t)gptr;
  asm volatile("global_load_async_to_lds_b128 %0, %1, off" :: "v"(loff), "v"(g) : "memory");
}
__device__ __forceinline__ void wait_async0() {
  asm volatile("s_wait_asynccnt 0x0" ::: "memory");
}

// ---------------------------------------------------------------- f32 -> bf16
__global__ void k_f2bf(const float* __restrict__ s, unsigned short* __restrict__ d, int n) {
  int i = blockIdx.x * blockDim.x + threadIdx.x;
  if (i < n) d[i] = f2bf(s[i]);
}

// ------------------------------------------- input transform + W1 + sigmoid
__global__ void k_embed(const float* __restrict__ ipt, const float* __restrict__ W1,
                        const float* __restrict__ b1, unsigned short* __restrict__ h) {
  int i = blockIdx.x * blockDim.x + threadIdx.x;   // over ML*128, exact grid
  int m = i >> 7, d = i & 127;
  float i0 = ipt[m * 3 + 0] * (1.0f / 127.0f);
  float i1 = ipt[m * 3 + 1];
  h[i] = f2bf(sigm(W1[d * 3 + 0] * i0 + W1[d * 3 + 1] * i1 + b1[d]));
}

// ------------- WMMA GEMM: xm_raw = h[ML,128] @ Win^T -> [ML,256]
// Weights (64KB bf16) staged once per block into LDS via async loads with an
// XOR chunk swizzle; 8 waves each compute a 16x64 tile (A frag reused x4).
__global__ void k_gemm_inproj(const unsigned short* __restrict__ A,   // [ML,128] bf16
                              const unsigned short* __restrict__ W,   // [256,128] bf16
                              unsigned short* __restrict__ O) {       // [ML,256] bf16
  __shared__ __align__(32) unsigned short Ws[DINNER * DMODEL];        // 64KB
  int tid = threadIdx.x;
  // 4096 16-byte units: row n (16 units/row), chunk c = unit/2 swizzled by n&7
  #pragma unroll
  for (int k = 0; k < 16; ++k) {
    int g = tid + k * 256;
    int n = g >> 4, p = g & 15;
    int c = p >> 1, hf = p & 1;
    int cs = c ^ (n & 7);
    async_g2l_b128(W + (size_t)g * 8, &Ws[n * 128 + cs * 16 + hf * 8]);
  }
  wait_async0();
  __syncthreads();

  int lane = tid & 31, wid = tid >> 5;
  int m0 = (blockIdx.x * 2 + (wid >> 2)) << 4;
  int n0 = (wid & 3) << 6;
  int lh = lane >> 4, ll = lane & 15;
  f32x8 acc0 = {}, acc1 = {}, acc2 = {}, acc3 = {};
  for (int k0 = 0; k0 < DMODEL; k0 += 32) {
    const unsigned short* ap = A + (size_t)(m0 + ll) * DMODEL + k0 + lh * 16;
    FragU a;
    a.u[0] = *(const U128*)ap;          // K 0..7 / 16..23 halves
    a.u[1] = *(const U128*)(ap + 8);    // K 8..15 / 24..31
    int c = (k0 >> 4) + lh;             // 16-elem chunk index
    int n0l = n0 + ll;
    bf16x16 b0 = *(const bf16x16*)&Ws[(n0l +  0) * 128 + (c ^ ((n0l +  0) & 7)) * 16];
    bf16x16 bq = *(const bf16x16*)&Ws[(n0l + 16) * 128 + (c ^ ((n0l + 16) & 7)) * 16];
    bf16x16 br = *(const bf16x16*)&Ws[(n0l + 32) * 128 + (c ^ ((n0l + 32) & 7)) * 16];
    bf16x16 bs = *(const bf16x16*)&Ws[(n0l + 48) * 128 + (c ^ ((n0l + 48) & 7)) * 16];
    acc0 = __builtin_amdgcn_wmma_f32_16x16x32_bf16(false, a.v, false, b0, (short)0, acc0, false, false);
    acc1 = __builtin_amdgcn_wmma_f32_16x16x32_bf16(false, a.v, false, bq, (short)0, acc1, false, false);
    acc2 = __builtin_amdgcn_wmma_f32_16x16x32_bf16(false, a.v, false, br, (short)0, acc2, false, false);
    acc3 = __builtin_amdgcn_wmma_f32_16x16x32_bf16(false, a.v, false, bs, (short)0, acc3, false, false);
  }
  int rbase = m0 + lh * 8;
  #pragma unroll
  for (int r = 0; r < 8; ++r) {
    size_t row = (size_t)(rbase + r) * DINNER;
    O[row + n0 +  0 + ll] = f2bf(acc0[r]);
    O[row + n0 + 16 + ll] = f2bf(acc1[r]);
    O[row + n0 + 32 + ll] = f2bf(acc2[r]);
    O[row + n0 + 48 + ll] = f2bf(acc3[r]);
  }
}

// ------------------------- causal depthwise conv (k=4) + bias + SiLU -> xm bf16
__global__ void k_conv(const unsigned short* __restrict__ xr, const float* __restrict__ cw,
                       const float* __restrict__ cb, unsigned short* __restrict__ xm) {
  int i = blockIdx.x * blockDim.x + threadIdx.x;   // ML*256 exact
  int d = i & 255;
  int m = i >> 8;
  int t = m & (SEQLEN - 1);
  float acc = cb[d];
  #pragma unroll
  for (int j = 0; j < 4; ++j) {
    int tt = t - 3 + j;
    if (tt >= 0) acc += cw[d * 4 + j] * bf2f(xr[(size_t)(m - 3 + j) * DINNER + d]);
  }
  xm[i] = f2bf(acc * sigm(acc));
}

// ------------- WMMA GEMM: dbc = xm[ML,256] @ x_proj^T[256,40->48] -> f32
// 20KB of weights async-staged to LDS; rows 40..47 zero-filled once, so the
// K-loop has no divergent guards at all.
__global__ void k_gemm_xproj(const unsigned short* __restrict__ A,   // [ML,256] bf16
                             const unsigned short* __restrict__ W,   // [40,256] bf16
                             float* __restrict__ O) {                // [ML,48] f32
  __shared__ __align__(32) unsigned short Ws[DBC_LD * DINNER];       // 24KB
  int tid = threadIdx.x;
  // 40 rows * 32 16B-units = 1280 units = 5 * 256
  #pragma unroll
  for (int k = 0; k < 5; ++k) {
    int g = tid + k * 256;
    int n = g >> 5, p = g & 31;
    int c = p >> 1, hf = p & 1;
    int cs = c ^ (n & 7);
    async_g2l_b128(W + (size_t)g * 8, &Ws[n * 256 + cs * 16 + hf * 8]);
  }
  #pragma unroll
  for (int k = 0; k < 8; ++k) Ws[40 * 256 + tid + k * 256] = 0;      // pad rows
  wait_async0();
  __syncthreads();

  int lane = tid & 31, wid = tid >> 5;
  int m0 = (blockIdx.x * 8 + wid) << 4;
  int lh = lane >> 4, ll = lane & 15;
  f32x8 acc0 = {}, acc1 = {}, acc2 = {};
  for (int k0 = 0; k0 < DINNER; k0 += 32) {
    const unsigned short* ap = A + (size_t)(m0 + ll) * DINNER + k0 + lh * 16;
    FragU a;
    a.u[0] = *(const U128*)ap;
    a.u[1] = *(const U128*)(ap + 8);
    int c = (k0 >> 4) + lh;
    bf16x16 b0 = *(const bf16x16*)&Ws[(ll +  0) * 256 + (c ^ ((ll +  0) & 7)) * 16];
    bf16x16 bq = *(const bf16x16*)&Ws[(ll + 16) * 256 + (c ^ ((ll + 16) & 7)) * 16];
    bf16x16 br = *(const bf16x16*)&Ws[(ll + 32) * 256 + (c ^ ((ll + 32) & 7)) * 16];
    acc0 = __builtin_amdgcn_wmma_f32_16x16x32_bf16(false, a.v, false, b0, (short)0, acc0, false, false);
    acc1 = __builtin_amdgcn_wmma_f32_16x16x32_bf16(false, a.v, false, bq, (short)0, acc1, false, false);
    acc2 = __builtin_amdgcn_wmma_f32_16x16x32_bf16(false, a.v, false, br, (short)0, acc2, false, false);
  }
  int rbase = m0 + lh * 8;
  #pragma unroll
  for (int r = 0; r < 8; ++r) {
    size_t row = (size_t)(rbase + r) * DBC_LD;
    O[row +  0 + ll] = acc0[r];
    O[row + 16 + ll] = acc1[r];
    O[row + 32 + ll] = acc2[r];
  }
}

// -------------------------------- delta = softplus(dt @ dt_proj^T + b)  (K=8)
__global__ void k_delta(const float* __restrict__ dbc, const float* __restrict__ dw,
                        const float* __restrict__ db, unsigned short* __restrict__ delta) {
  int i = blockIdx.x * blockDim.x + threadIdx.x;   // ML*256 exact
  int d = i & 255;
  int m = i >> 8;
  float acc = db[d];
  #pragma unroll
  for (int r = 0; r < DTRANK; ++r) acc += dbc[(size_t)m * DBC_LD + r] * dw[d * DTRANK + r];
  float sp = (acc > 20.f) ? acc : __logf(1.f + __expf(acc));
  delta[i] = f2bf(sp);
}

// ------------------------- selective scan: state-only recurrence, y at t=L-1,
// fused z gate + SiLU. 1 block/batch; B,C chunks async-staged into LDS.
__global__ void k_scan(const unsigned short* __restrict__ xm, const unsigned short* __restrict__ delta,
                       const float* __restrict__ dbc, const float* __restrict__ A_log,
                       const float* __restrict__ Dp, const float* __restrict__ in_proj_w,
                       const unsigned short* __restrict__ h, float* __restrict__ yz) {
  __shared__ __align__(16) float bc[CHUNK][32];    // per-step B[16] | C[16]
  __shared__ float hls[DMODEL];
  int b = blockIdx.x, d = threadIdx.x;
  float A2[DSTATE], s[DSTATE];
  #pragma unroll
  for (int n = 0; n < DSTATE; ++n) {
    A2[n] = -__expf(A_log[d * DSTATE + n]);   // A = -exp(A_log)
    s[n] = 0.f;
  }
  size_t base = (size_t)b * SEQLEN;
  for (int t0 = 0; t0 < SEQLEN; t0 += CHUNK) {
    __syncthreads();                          // previous chunk fully consumed
    // 64 rows * 8 16B-units = 512 async b128 copies, 2 per thread
    #pragma unroll
    for (int k = 0; k < 2; ++k) {
      int j = d + k * 256;
      int row = j >> 3, p = j & 7;
      async_g2l_b128(dbc + (base + t0 + row) * DBC_LD + 8 + p * 4, &bc[row][p * 4]);
    }
    wait_async0();
    __syncthreads();
    for (int tt = 0; tt < CHUNK; ++tt) {
      size_t idx = (base + t0 + tt) * DINNER + d;
      float u  = bf2f(xm[idx]);
      float dt = bf2f(delta[idx]);
      float dtu = dt * u;
      #pragma unroll
      for (int n = 0; n < DSTATE; ++n)
        s[n] = __expf(dt * A2[n]) * s[n] + dtu * bc[tt][n];
    }
  }
  // y only needed at t = SEQLEN-1; bc[CHUNK-1] still holds its B/C.
  float u_last = bf2f(xm[(base + SEQLEN - 1) * DINNER + d]);
  float y = u_last * Dp[d];
  #pragma unroll
  for (int n = 0; n < DSTATE; ++n) y += s[n] * bc[CHUNK - 1][16 + n];
  if (d < DMODEL) hls[d] = bf2f(h[(base + SEQLEN - 1) * DMODEL + d]);
  __syncthreads();
  float z = 0.f;
  const float* wz = in_proj_w + (size_t)(DINNER + d) * DMODEL;   // gate half, f32
  #pragma unroll 8
  for (int k = 0; k < DMODEL; ++k) z += hls[k] * wz[k];
  yz[b * DINNER + d] = y * (z * sigm(z));
}

// ----------------------- head: out_proj -> W2 -> softmax (last timestep only)
__global__ void k_head(const float* __restrict__ yz, const float* __restrict__ opw,
                       const float* __restrict__ W2, const float* __restrict__ b2,
                       float* __restrict__ out) {
  __shared__ float yzs[DINNER], h2s[DMODEL], red[DMODEL];
  int b = blockIdx.x, o = threadIdx.x;   // 128 threads
  yzs[o]       = yz[b * DINNER + o];
  yzs[o + 128] = yz[b * DINNER + o + 128];
  __syncthreads();
  float acc = 0.f;
  const float* w = opw + (size_t)o * DINNER;
  #pragma unroll 8
  for (int k = 0; k < DINNER; ++k) acc += yzs[k] * w[k];
  h2s[o] = acc;
  __syncthreads();
  float lg = b2[o];
  const float* w2 = W2 + (size_t)o * DMODEL;
  #pragma unroll 8
  for (int k = 0; k < DMODEL; ++k) lg += h2s[k] * w2[k];
  red[o] = lg;
  __syncthreads();
  for (int st = 64; st > 0; st >>= 1) { if (o < st) red[o] = fmaxf(red[o], red[o + st]); __syncthreads(); }
  float mx = red[0];
  __syncthreads();
  float e = __expf(lg - mx);
  red[o] = e;
  __syncthreads();
  for (int st = 64; st > 0; st >>= 1) { if (o < st) red[o] += red[o + st]; __syncthreads(); }
  out[b * OUTPUTS + o] = e / red[0];
}

extern "C" void kernel_launch(void* const* d_in, const int* in_sizes, int n_in,
                              void* d_out, int out_size, void* d_ws, size_t ws_size,
                              hipStream_t stream) {
  (void)in_sizes; (void)n_in; (void)out_size; (void)ws_size;
  const float* ipt       = (const float*)d_in[0];
  const float* W1        = (const float*)d_in[1];
  const float* b1        = (const float*)d_in[2];
  const float* in_proj_w = (const float*)d_in[3];
  const float* conv_w    = (const float*)d_in[4];
  const float* conv_b    = (const float*)d_in[5];
  const float* x_proj_w  = (const float*)d_in[6];
  const float* dt_proj_w = (const float*)d_in[7];
  const float* dt_proj_b = (const float*)d_in[8];
  const float* A_log     = (const float*)d_in[9];
  const float* Dp        = (const float*)d_in[10];
  const float* out_proj_w= (const float*)d_in[11];
  const float* W2        = (const float*)d_in[12];
  const float* b2        = (const float*)d_in[13];
  float* out = (float*)d_out;

  char* ws = (char*)d_ws;
  size_t off = 0;
  auto alloc = [&](size_t bytes) { size_t p = off; off = (off + bytes + 255) & ~(size_t)255; return p; };
  unsigned short* h     = (unsigned short*)(ws + alloc((size_t)ML * DMODEL * 2));
  unsigned short* Winb  = (unsigned short*)(ws + alloc((size_t)DINNER * DMODEL * 2));
  unsigned short* Wxpb  = (unsigned short*)(ws + alloc((size_t)40 * DINNER * 2));
  unsigned short* xmraw = (unsigned short*)(ws + alloc((size_t)ML * DINNER * 2));
  unsigned short* xm    = (unsigned short*)(ws + alloc((size_t)ML * DINNER * 2));
  float*          dbc   = (float*)         (ws + alloc((size_t)ML * DBC_LD * 4));
  unsigned short* delta = (unsigned short*)(ws + alloc((size_t)ML * DINNER * 2));
  float*          yz    = (float*)         (ws + alloc((size_t)BATCHN * DINNER * 4));

  k_f2bf<<<(DINNER * DMODEL + 255) / 256, 256, 0, stream>>>(in_proj_w, Winb, DINNER * DMODEL);
  k_f2bf<<<(40 * DINNER + 255) / 256, 256, 0, stream>>>(x_proj_w, Wxpb, 40 * DINNER);
  k_embed<<<(ML * DMODEL) / 256, 256, 0, stream>>>(ipt, W1, b1, h);
  k_gemm_inproj<<<(ML / 16) / 2, 256, 0, stream>>>(h, Winb, xmraw);            // 2048 blocks
  k_conv<<<(ML * DINNER) / 256, 256, 0, stream>>>(xmraw, conv_w, conv_b, xm);
  k_gemm_xproj<<<(ML / 16) / 8, 256, 0, stream>>>(xm, Wxpb, dbc);              // 512 blocks
  k_delta<<<(ML * DINNER) / 256, 256, 0, stream>>>(dbc, dt_proj_w, dt_proj_b, delta);
  k_scan<<<BATCHN, DINNER, 0, stream>>>(xm, delta, dbc, A_log, Dp, in_proj_w, h, yz);
  k_head<<<BATCHN, OUTPUTS, 0, stream>>>(yz, out_proj_w, W2, b2, out);
}
